// Decoder_45260365365566
// MI455X (gfx1250) — matrix operand order repaired and checked by reading, not compiled, not run on hardware
//
#include <hip/hip_runtime.h>
#include <hip/hip_bf16.h>

typedef __attribute__((ext_vector_type(16))) _Float16 v16h;
typedef __attribute__((ext_vector_type(8)))  _Float16 v8h;
typedef __attribute__((ext_vector_type(8)))  float    v8f;

#define BB 2
#define LL 2048
#define NN 4096
#define DD 64
#define OO 128
#define FTCLIP 0.2f

// ---------------- workspace layout ----------------
// float region
static constexpr size_t F_GT   = 0;              // [B][64] goal_trans
static constexpr size_t F_GN   = 128;            // [B] goal norms (padded)
static constexpr size_t F_WOP  = 136;            // [B][O][64] word_op
static constexpr size_t F_TOT  = F_WOP + (size_t)BB * OO * DD;   // floats
static constexpr size_t H_BYTE = ((F_TOT * 4 + 255) / 256) * 256; // half region byte offset
// half region (units of _Float16, from half base)
static constexpr size_t HBLK   = (size_t)BB * NN * DD;  // 524288
static constexpr size_t H_ALLW = 0 * HBLK;   // all_w f16, row-major [B][N][64]
static constexpr size_t H_WG   = 1 * HBLK;   // word_g f16
static constexpr size_t H_XKT  = 2 * HBLK;   // (all_w@Wwk+bwk)^T  [B][64][N]
static constexpr size_t H_XST  = 3 * HBLK;   // (all_w@Wws+bws)^T  [B][64][N]
static constexpr size_t H_WWK  = 4 * HBLK;   // word_w_k  [B][N][64]
static constexpr size_t H_WWS  = 5 * HBLK;   // word_w_s  [B][N][64]
static constexpr size_t H_WU   = 6 * HBLK;   // word_updated f16
static constexpr size_t H_XOT  = 7 * HBLK;   // (wu@Wo+bo)^T [B][64][N]
static constexpr size_t H_W    = 8 * HBLK;   // transposed f16 weights
static constexpr size_t H_WWKT = H_W + 0;        // [64][64]
static constexpr size_t H_WWST = H_W + 4096;     // [64][64]
static constexpr size_t H_WOT  = H_W + 8192;     // [64][64]
static constexpr size_t H_WFT  = H_W + 12288;    // [64][256]
static constexpr size_t H_WUPT = H_W + 28672;    // [64][192]

// ---------------- WMMA fragment helpers ----------------
static __device__ __forceinline__ v16h pack2(v8h lo, v8h hi) {
  v16h r;
#pragma unroll
  for (int i = 0; i < 8; i++) { r[i] = lo[i]; r[i + 8] = hi[i]; }
  return r;
}
// 16-bit A fragment (16x32): lane m=l&15, half=l>>4 -> K runs [8h,8h+8) and [16+8h,24+8h)
static __device__ __forceinline__ v16h fragA_h(const _Float16* row, int kbase, int half) {
  v8h lo = *(const v8h*)(row + kbase + 8 * half);
  v8h hi = *(const v8h*)(row + kbase + 16 + 8 * half);
  return pack2(lo, hi);
}
// 16-bit B fragment (32x16) from transposed storage BT[n][k]: K runs [8h,8h+8) and [16+8h,24+8h)
static __device__ __forceinline__ v16h fragB_h(const _Float16* bt_row, int kbase, int half) {
  const _Float16* p = bt_row + kbase + 8 * half;
  return pack2(*(const v8h*)p, *(const v8h*)(p + 16));
}
static __device__ __forceinline__ v8f wmma16(v16h a, v16h b, v8f c) {
  return __builtin_amdgcn_wmma_f32_16x16x32_f16(false, a, false, b, (short)0, c, false, false);
}
static __device__ __forceinline__ v8f vzero() {
  v8f z = {0.f, 0.f, 0.f, 0.f, 0.f, 0.f, 0.f, 0.f};
  return z;
}

// ---------------- small kernels ----------------
__global__ void k_transpose_w(const float* __restrict__ src, _Float16* __restrict__ dst, int rows) {
  int idx = blockIdx.x * blockDim.x + threadIdx.x;
  if (idx >= rows * 64) return;
  int i = idx >> 6, o = idx & 63;
  dst[(size_t)o * rows + i] = (_Float16)src[(size_t)i * 64 + o];
}

__global__ void k_goal(const float* __restrict__ nh, const float* __restrict__ Wg,
                       const float* __restrict__ bg, float* __restrict__ gt,
                       float* __restrict__ gn) {
  int b = blockIdx.x, d = threadIdx.x;
  float s = bg[d];
  for (int k = 0; k < 64; k++) s += nh[b * 64 + k] * Wg[k * 64 + d];
  gt[b * 64 + d] = s;
  __shared__ float sh[64];
  sh[d] = s * s;
  __syncthreads();
  if (d == 0) {
    float t = 0.f;
    for (int i = 0; i < 64; i++) t += sh[i];
    gn[b] = sqrtf(t);
  }
}

__global__ void k_elem(const float* __restrict__ w0, const float* __restrict__ w1,
                       const float* __restrict__ gw, const float* __restrict__ gt,
                       const float* __restrict__ gn, _Float16* __restrict__ allw,
                       _Float16* __restrict__ wg) {
  int idx = blockIdx.x * blockDim.x + threadIdx.x;  // B*N*64
  int b = idx >> 18, rem = idx & 262143, n = rem >> 6, d = rem & 63;
  float a = (n < LL) ? w0[((size_t)b * LL + n) * DD + d]
                     : w1[((size_t)b * LL + (n - LL)) * DD + d];
  allw[idx] = (_Float16)a;
  float c = gw[(size_t)b * NN + n];
  float g = gt[b * 64 + d];
  wg[idx] = (_Float16)(g * c / (gn[b] * c + 1e-30f));  // == l2n(gt*c) row-wise
}

// ---------------- projection GEMM: out^T = (A[Nx64] @ W[64x64] + b)^T ----------------
__global__ void __launch_bounds__(128) k_proj(const _Float16* __restrict__ A,
                                              const _Float16* __restrict__ WT1,
                                              const float* __restrict__ b1,
                                              _Float16* __restrict__ o1,
                                              const _Float16* __restrict__ WT2,
                                              const float* __restrict__ b2,
                                              _Float16* __restrict__ o2) {
  int blk = blockIdx.x, b = blk >> 6, rb = blk & 63;
  int tid = threadIdx.x, wid = tid >> 5, l = tid & 31, m = l & 15, half = l >> 4;
  int rowbase = rb * 64 + wid * 16;
  const _Float16* arow = A + ((size_t)b * NN + rowbase + m) * DD;
  v16h a0 = fragA_h(arow, 0, half);
  v16h a1 = fragA_h(arow, 32, half);
  const _Float16* WTs[2] = {WT1, WT2};
  const float* bs[2] = {b1, b2};
  _Float16* os[2] = {o1, o2};
  int np = WT2 ? 2 : 1;
  for (int p = 0; p < np; p++) {
#pragma unroll
    for (int t = 0; t < 4; t++) {
      int col = 16 * t + m;
      v8f acc = vzero();
      acc = wmma16(a0, fragB_h(WTs[p] + (size_t)col * 64, 0, half), acc);
      acc = wmma16(a1, fragB_h(WTs[p] + (size_t)col * 64, 32, half), acc);
      float bias = bs[p][col];
      v8h st;
#pragma unroll
      for (int r = 0; r < 8; r++) st[r] = (_Float16)(acc[r] + bias);
      *(v8h*)(os[p] + ((size_t)b * DD + col) * NN + rowbase + 8 * half) = st;
    }
  }
}

// ---------------- word_w_k: l2n((ww*wem) @ Xk) ----------------
__global__ void __launch_bounds__(128) k_gemm_mask(const float* __restrict__ ww,
                                                   const float* __restrict__ wem,
                                                   const _Float16* __restrict__ XkT,
                                                   _Float16* __restrict__ out) {
  int blk = blockIdx.x, b = blk >> 6, rb = blk & 63;
  int tid = threadIdx.x, wid = tid >> 5, l = tid & 31, m = l & 15, half = l >> 4;
  int rowbase = rb * 64 + wid * 16, row = rowbase + m;
  const float* pw = ww + ((size_t)b * NN + row) * NN;
  const float* pm = wem + ((size_t)b * NN + row) * NN;
  v8f acc[4] = {vzero(), vzero(), vzero(), vzero()};
  for (int kb = 0; kb < NN; kb += 32) {
    if (kb + 96 < NN) {
      __builtin_prefetch(pw + kb + 96, 0, 0);
      __builtin_prefetch(pm + kb + 96, 0, 0);
    }
    v16h a;
    int k0 = kb + 8 * half, k1 = kb + 16 + 8 * half;
#pragma unroll
    for (int i = 0; i < 8; i++) a[i] = (_Float16)(pw[k0 + i] * pm[k0 + i]);
#pragma unroll
    for (int i = 0; i < 8; i++) a[8 + i] = (_Float16)(pw[k1 + i] * pm[k1 + i]);
#pragma unroll
    for (int t = 0; t < 4; t++)
      acc[t] = wmma16(a, fragB_h(XkT + ((size_t)b * DD + 16 * t + m) * NN, kb, half), acc[t]);
  }
  float inv[8];
#pragma unroll
  for (int r = 0; r < 8; r++) {
    float s = 0.f;
#pragma unroll
    for (int t = 0; t < 4; t++) { float v = acc[t][r]; s += v * v; }
    s += __shfl_xor(s, 1, 32); s += __shfl_xor(s, 2, 32);
    s += __shfl_xor(s, 4, 32); s += __shfl_xor(s, 8, 32);
    inv[r] = 1.f / (sqrtf(s) + 1e-30f);
  }
#pragma unroll
  for (int t = 0; t < 4; t++)
#pragma unroll
    for (int r = 0; r < 8; r++)
      out[((size_t)b * NN + rowbase + 8 * half + r) * DD + 16 * t + m] =
          (_Float16)(acc[t][r] * inv[r]);
}

// ---------------- word_w_s: l2n(depend @ Xs) ----------------
__global__ void __launch_bounds__(128) k_gemm_dep(const float* __restrict__ d0,
                                                  const float* __restrict__ d1,
                                                  const _Float16* __restrict__ XsT,
                                                  _Float16* __restrict__ out) {
  int blk = blockIdx.x, b = blk >> 6, rb = blk & 63;
  int tid = threadIdx.x, wid = tid >> 5, l = tid & 31, m = l & 15, half = l >> 4;
  int rowbase = rb * 64 + wid * 16, row = rowbase + m;
  int lower = (rowbase < LL);
  const float* dm = lower ? d0 : d1;
  int rl = row - (lower ? 0 : LL);
  int coff = lower ? 0 : LL;
  const float* pr = dm + ((size_t)b * LL + rl) * LL;
  v8f acc[4] = {vzero(), vzero(), vzero(), vzero()};
  for (int kb = 0; kb < LL; kb += 32) {
    if (kb + 96 < LL) __builtin_prefetch(pr + kb + 96, 0, 0);
    v16h a;
    int k0 = kb + 8 * half, k1 = kb + 16 + 8 * half;
#pragma unroll
    for (int i = 0; i < 8; i++) a[i] = (_Float16)pr[k0 + i];
#pragma unroll
    for (int i = 0; i < 8; i++) a[8 + i] = (_Float16)pr[k1 + i];
#pragma unroll
    for (int t = 0; t < 4; t++)
      acc[t] = wmma16(a, fragB_h(XsT + ((size_t)b * DD + 16 * t + m) * NN + coff, kb, half), acc[t]);
  }
  float inv[8];
#pragma unroll
  for (int r = 0; r < 8; r++) {
    float s = 0.f;
#pragma unroll
    for (int t = 0; t < 4; t++) { float v = acc[t][r]; s += v * v; }
    s += __shfl_xor(s, 1, 32); s += __shfl_xor(s, 2, 32);
    s += __shfl_xor(s, 4, 32); s += __shfl_xor(s, 8, 32);
    inv[r] = 1.f / (sqrtf(s) + 1e-30f);
  }
#pragma unroll
  for (int t = 0; t < 4; t++)
#pragma unroll
    for (int r = 0; r < 8; r++)
      out[((size_t)b * NN + rowbase + 8 * half + r) * DD + 16 * t + m] =
          (_Float16)(acc[t][r] * inv[r]);
}

// ---------------- fused forget/update ----------------
__global__ void __launch_bounds__(128) k_fused(const _Float16* __restrict__ allw,
                                               const _Float16* __restrict__ wg,
                                               const _Float16* __restrict__ wwk,
                                               const _Float16* __restrict__ wws,
                                               const _Float16* __restrict__ WfT,
                                               const float* __restrict__ bf,
                                               const _Float16* __restrict__ WupdT,
                                               const float* __restrict__ bupd,
                                               const float* __restrict__ w0,
                                               const float* __restrict__ w1,
                                               float* __restrict__ dout,
                                               _Float16* __restrict__ wu) {
  int blk = blockIdx.x, b = blk >> 6, rb = blk & 63;
  int tid = threadIdx.x, wid = tid >> 5, l = tid & 31, m = l & 15, half = l >> 4;
  int rowbase = rb * 64 + wid * 16, row = rowbase + m;
  const _Float16* segs[4] = {allw, wg, wwk, wws};
  v8f F[4] = {vzero(), vzero(), vzero(), vzero()};
  v8f U[4] = {vzero(), vzero(), vzero(), vzero()};
#pragma unroll
  for (int ks = 0; ks < 8; ks++) {
    const _Float16* arow = segs[ks >> 1] + ((size_t)b * NN + row) * DD;
    v16h a = fragA_h(arow, (ks & 1) * 32, half);
#pragma unroll
    for (int t = 0; t < 4; t++) {
      int col = 16 * t + m;
      F[t] = wmma16(a, fragB_h(WfT + (size_t)col * 256, ks * 32, half), F[t]);
      if (ks >= 2)
        U[t] = wmma16(a, fragB_h(WupdT + (size_t)col * 192, (ks - 2) * 32, half), U[t]);
    }
  }
#pragma unroll
  for (int t = 0; t < 4; t++) {
    int col = 16 * t + m;
#pragma unroll
    for (int r = 0; r < 8; r++) {
      int rr = rowbase + 8 * half + r;
      float f = 1.f / (1.f + __expf(-(F[t][r] + bf[col])));
      float u = fmaxf(U[t][r] + bupd[col], 0.f);
      float aw = (rr < LL) ? w0[((size_t)b * LL + rr) * DD + col]
                           : w1[((size_t)b * LL + (rr - LL)) * DD + col];
      float v = fmaxf(f, FTCLIP) * aw + (1.f - f) * u;
      dout[((size_t)b * (NN + OO) + rr) * DD + col] = v;
      wu[((size_t)b * NN + rr) * DD + col] = (_Float16)v;
    }
  }
}

// ---------------- word_op = norm_mat((wes*wop)^T) @ Xo ----------------
__global__ void __launch_bounds__(128) k_gemm_op(const float* __restrict__ wes,
                                                 const float* __restrict__ wop,
                                                 const _Float16* __restrict__ XoT,
                                                 float* __restrict__ wopbuf) {
  int b = blockIdx.x >> 1, rb = blockIdx.x & 1;
  int tid = threadIdx.x, wid = tid >> 5, l = tid & 31, m = l & 15, half = l >> 4;
  int base = rb * 64 + wid * 16, o = base + m;
  const float* pe = wes + (size_t)b * NN;
  const float* po = wop + (size_t)b * NN * OO + o;
  v8f acc[4] = {vzero(), vzero(), vzero(), vzero()};
  float deg = 0.f;
  for (int kb = 0; kb < NN; kb += 32) {
    v16h a;
    int k0 = kb + 8 * half, k1 = kb + 16 + 8 * half;
#pragma unroll
    for (int i = 0; i < 8; i++) {
      float v = pe[k0 + i] * po[(size_t)(k0 + i) * OO];
      deg += v; a[i] = (_Float16)v;
    }
#pragma unroll
    for (int i = 0; i < 8; i++) {
      float v = pe[k1 + i] * po[(size_t)(k1 + i) * OO];
      deg += v; a[8 + i] = (_Float16)v;
    }
#pragma unroll
    for (int t = 0; t < 4; t++)
      acc[t] = wmma16(a, fragB_h(XoT + ((size_t)b * DD + 16 * t + m) * NN, kb, half), acc[t]);
  }
  deg += __shfl_xor(deg, 16, 32);  // combine K-halves: row degree (binary mask -> count)
  __shared__ float sdeg[4][16];
  if (l < 16) sdeg[wid][l] = deg;
  __syncthreads();
#pragma unroll
  for (int t = 0; t < 4; t++)
#pragma unroll
    for (int r = 0; r < 8; r++) {
      float dgg = sdeg[wid][r + 8 * half];
      wopbuf[((size_t)b * OO + base + 8 * half + r) * DD + 16 * t + m] =
          acc[t][r] / (dgg + 1e-30f);
    }
}

// ---------------- op output (tiny: B*O rows) ----------------
__global__ void k_opout(const float* __restrict__ opE, const float* __restrict__ wopbuf,
                        const float* __restrict__ Wf2, const float* __restrict__ bf2,
                        const float* __restrict__ Wout, const float* __restrict__ bout,
                        float* __restrict__ dout) {
  int b = blockIdx.x >> 7, o = blockIdx.x & 127, d = threadIdx.x;
  __shared__ float sE[64], sW[64];
  sE[d] = opE[((size_t)b * OO + o) * DD + d];
  sW[d] = wopbuf[((size_t)b * OO + o) * DD + d];
  __syncthreads();
  float f = bf2[d], g = bout[d];
  for (int i = 0; i < 64; i++) f += sE[i] * Wf2[i * 64 + d];
  for (int i = 0; i < 64; i++) f += sW[i] * Wf2[(64 + i) * 64 + d];
  for (int i = 0; i < 64; i++) g += sW[i] * Wout[i * 64 + d];
  f = 1.f / (1.f + __expf(-f));
  float v = fmaxf(f, FTCLIP) * sE[d] + (1.f - f) * fmaxf(g, 0.f);
  dout[((size_t)b * (NN + OO) + NN + o) * DD + d] = v;
}

// ---------------- launch ----------------
extern "C" void kernel_launch(void* const* d_in, const int* in_sizes, int n_in,
                              void* d_out, int out_size, void* d_ws, size_t ws_size,
                              hipStream_t stream) {
  const float* w0   = (const float*)d_in[0];
  const float* w1   = (const float*)d_in[1];
  const float* nh   = (const float*)d_in[2];
  const float* opE  = (const float*)d_in[3];
  const float* wes  = (const float*)d_in[4];
  const float* wem  = (const float*)d_in[5];
  const float* wop  = (const float*)d_in[6];
  const float* ww   = (const float*)d_in[7];
  const float* dp0  = (const float*)d_in[8];
  const float* dp1  = (const float*)d_in[9];
  const float* gw   = (const float*)d_in[10];
  const float* Wg   = (const float*)d_in[11];
  const float* bg   = (const float*)d_in[12];
  const float* Wwk  = (const float*)d_in[13];
  const float* bwk  = (const float*)d_in[14];
  const float* Wws  = (const float*)d_in[15];
  const float* bws  = (const float*)d_in[16];
  const float* Wo   = (const float*)d_in[17];
  const float* bo   = (const float*)d_in[18];
  // Wupd=19, bupd=20, Wf=21, bf=22, Wf2=23, bf2=24, Wout=25, bout=26
  const float* Wupd = (const float*)d_in[19];
  const float* bupd = (const float*)d_in[20];
  const float* Wf   = (const float*)d_in[21];
  const float* bf   = (const float*)d_in[22];
  const float* Wf2  = (const float*)d_in[23];
  const float* bf2  = (const float*)d_in[24];
  const float* Wout = (const float*)d_in[25];
  const float* bout = (const float*)d_in[26];
  float* out = (float*)d_out;

  char* wsb = (char*)d_ws;
  float* fws = (float*)wsb;
  float* gt = fws + F_GT;
  float* gn = fws + F_GN;
  float* wopbuf = fws + F_WOP;
  _Float16* h = (_Float16*)(wsb + H_BYTE);

  // weight conversions (f16 transposed)
  k_transpose_w<<<16, 256, 0, stream>>>(Wwk, h + H_WWKT, 64);
  k_transpose_w<<<16, 256, 0, stream>>>(Wws, h + H_WWST, 64);
  k_transpose_w<<<16, 256, 0, stream>>>(Wo,  h + H_WOT,  64);
  k_transpose_w<<<64, 256, 0, stream>>>(Wf,  h + H_WFT,  256);
  k_transpose_w<<<48, 256, 0, stream>>>(Wupd, h + H_WUPT, 192);
  // goal vector + norm
  k_goal<<<BB, 64, 0, stream>>>(nh, Wg, bg, gt, gn);
  // all_w f16 + word_g f16
  k_elem<<<2048, 256, 0, stream>>>(w0, w1, gw, gt, gn, h + H_ALLW, h + H_WG);
  // projections Xk^T, Xs^T
  k_proj<<<128, 128, 0, stream>>>(h + H_ALLW, h + H_WWKT, bwk, h + H_XKT,
                                  h + H_WWST, bws, h + H_XST);
  // big sparse-mask GEMMs
  k_gemm_mask<<<128, 128, 0, stream>>>(ww, wem, h + H_XKT, h + H_WWK);
  k_gemm_dep<<<128, 128, 0, stream>>>(dp0, dp1, h + H_XST, h + H_WWS);
  // fused forget/update -> word_updated (d_out rows 0..N) + f16 copy
  k_fused<<<128, 128, 0, stream>>>(h + H_ALLW, h + H_WG, h + H_WWK, h + H_WWS,
                                   h + H_WFT, bf, h + H_WUPT, bupd, w0, w1, out, h + H_WU);
  // Xo^T = (wu @ Wo + bo)^T
  k_proj<<<128, 128, 0, stream>>>(h + H_WU, h + H_WOT, bo, h + H_XOT,
                                  nullptr, nullptr, nullptr);
  // word_op with fused degree normalization
  k_gemm_op<<<4, 128, 0, stream>>>(wes, wop, h + H_XOT, wopbuf);
  // op outputs -> d_out rows N..N+O
  k_opout<<<BB * OO, 64, 0, stream>>>(opE, wopbuf, Wf2, bf2, Wout, bout, out);
}
